// Decoder_Transformer_57123065036836
// MI455X (gfx1250) — matrix operand order repaired and checked by reading, not compiled, not run on hardware
//
#include <hip/hip_runtime.h>
#include <hip/hip_bf16.h>

// Model dims (from reference): L=6, B=6, S=7, D=512, H=8, FF=2048, HD=64
#define L_ 6
#define B_ 6
#define S_ 7
#define D_ 512
#define H_ 8
#define FF_ 2048
#define HD_ 64

typedef __attribute__((ext_vector_type(16))) __bf16 v16bf;
typedef __attribute__((ext_vector_type(8)))  float  v8f;

// Convert 8 floats (two float4s) to bf16 into dst[O..O+7]
template <int O>
__device__ __forceinline__ void cvt8(v16bf& d, float4 x, float4 y) {
  d[O + 0] = (__bf16)x.x; d[O + 1] = (__bf16)x.y;
  d[O + 2] = (__bf16)x.z; d[O + 3] = (__bf16)x.w;
  d[O + 4] = (__bf16)y.x; d[O + 5] = (__bf16)y.y;
  d[O + 6] = (__bf16)y.z; d[O + 7] = (__bf16)y.w;
}

// ---------------------------------------------------------------------------
// Split-K WMMA GEMM over PADDED matrices:  part[ks] = A(Mp x K) * W(N x K)^T
// Mp = MT*16 rows, no validity masks anywhere: pad rows are garbage-in ->
// garbage-out on rows no consumer reads. All loads are unconditional float4
// (global_load_b128 clauses); W is streamed exactly once across the grid.
// ---------------------------------------------------------------------------
template <int MT>
__global__ void __launch_bounds__(32) wmma_gemm_splitk(
    const float* __restrict__ A, int lda,
    const float* __restrict__ W, int ldw,
    float* __restrict__ part,          // [KS][MT*16][N]
    int N, int K, int kchunk)
{
  const int lane = threadIdx.x & 31;
  const int r    = lane & 15;
  const int hv   = lane >> 4;          // lane half selects K sub-block
  const int tn   = blockIdx.x;         // N tile
  const int ks   = blockIdx.y;         // K chunk
  const int col  = tn * 16 + r;
  const float* Wrow = W + (size_t)col * ldw;   // B[k][n] = W[n][k]

  const float* Ar[MT];
#pragma unroll
  for (int t = 0; t < MT; ++t) Ar[t] = A + (size_t)(t * 16 + r) * lda;

  const int k_begin = ks * kchunk;
  int k_end = k_begin + kchunk; if (k_end > K) k_end = K;

  v8f acc[MT] = {};

  for (int k0 = k_begin; k0 < k_end; k0 += 32) {
    // B fragment (32x16): lane holds 16 consecutive K (k0 + 16*half .. +15)
    const float4* Bp = (const float4*)(Wrow + k0 + 16 * hv);
    float4 b0 = Bp[0], b1 = Bp[1], b2 = Bp[2], b3 = Bp[3];
    v16bf bfrag;
    cvt8<0>(bfrag, b0, b1);
    cvt8<8>(bfrag, b2, b3);
#pragma unroll
    for (int t = 0; t < MT; ++t) {
      // A fragment (16x32): lane holds 16 consecutive K (k0 + 16*half .. +15)
      const float4* Ap0 = (const float4*)(Ar[t] + k0 + 16 * hv);
      const float4* Ap1 = (const float4*)(Ar[t] + k0 + 16 * hv + 8);
      float4 a0 = Ap0[0], a1 = Ap0[1];
      float4 a2 = Ap1[0], a3 = Ap1[1];
      v16bf afrag;
      cvt8<0>(afrag, a0, a1);
      cvt8<8>(afrag, a2, a3);
      acc[t] = __builtin_amdgcn_wmma_f32_16x16x32_bf16(
          false, afrag, false, bfrag, (short)0, acc[t], false, false);
    }
  }

  // C/D layout: col = lane%16 ; row = v + 8*(lane>>4)
  float* pc = part + (size_t)ks * (MT * 16) * N;
#pragma unroll
  for (int t = 0; t < MT; ++t) {
#pragma unroll
    for (int v = 0; v < 8; ++v) {
      const int row = t * 16 + v + 8 * hv;
      pc[(size_t)row * N + col] = acc[t][v];
    }
  }
}

// Reduce split-K partials, add bias, optional activation. C is padded Mp x N.
// Optional fused gather: verbg row 0 collects C rows {7,14,21,28,35}
// (msg[b][0][:] for b=1..5) so no separate pack kernel is needed.
__global__ void __launch_bounds__(256) reduce_act_kernel(
    const float* __restrict__ part, const float* __restrict__ bias,
    float* __restrict__ C, int MN, int N, int KS, int act,
    float* __restrict__ verbg)
{
  const int i = blockIdx.x * blockDim.x + threadIdx.x;
  if (i >= MN) return;
  float s = 0.0f;
  for (int k = 0; k < KS; ++k) s += part[(size_t)k * MN + i];
  s += bias[i % N];
  if (act == 1)      s = fmaxf(s, 0.0f);
  else if (act == 2) s = 1.0f / (1.0f + expf(-s));
  C[i] = s;
  if (verbg) {
    const int row = i / N;
    if (row >= S_ && row < B_ * S_ && (row % S_) == 0)
      verbg[(size_t)(row / S_ - 1) * N + (i - row * N)] = s;
  }
}

// src = x ; src[:,1:,:] += role_tokens   (writes rows 0..41 of padded src48)
__global__ void __launch_bounds__(256) add_role_kernel(
    float* __restrict__ src, const float* __restrict__ x,
    const float* __restrict__ role)
{
  const int i = blockIdx.x * blockDim.x + threadIdx.x;
  const int total = B_ * S_ * D_;
  if (i >= total) return;
  const int b = i / (S_ * D_);
  const int rem = i - b * (S_ * D_);
  const int s = rem / D_;
  const int d = rem - s * D_;
  float v = x[i];
  if (s >= 1) v += role[((size_t)b * (S_ - 1) + (s - 1)) * D_ + d];
  src[i] = v;
}

// Per-(b,h) 7x7 attention on fp32 qkv (rows b*7+s, stride 3D); block = 512
__global__ void __launch_bounds__(512) attn_kernel(
    const float* __restrict__ qkv, float* __restrict__ ctx)
{
  __shared__ float sc[S_][S_];
  __shared__ float att[S_][S_];
  const int bh = blockIdx.x;
  const int b = bh / H_;
  const int h = bh % H_;
  const float* base = qkv + (size_t)b * S_ * (3 * D_);
  const int tid = threadIdx.x;

  if (tid < S_ * S_) {
    const int qi = tid / S_, ki = tid % S_;
    const float* qp = base + qi * (3 * D_) + h * HD_;
    const float* kp = base + ki * (3 * D_) + D_ + h * HD_;
    float sum = 0.0f;
    for (int d = 0; d < HD_; ++d) sum += qp[d] * kp[d];
    sc[qi][ki] = sum * 0.125f;              // 1/sqrt(64)
  }
  __syncthreads();
  if (tid < S_) {
    float m = sc[tid][0];
    for (int k = 1; k < S_; ++k) m = fmaxf(m, sc[tid][k]);
    float e[S_]; float s = 0.0f;
    for (int k = 0; k < S_; ++k) { e[k] = expf(sc[tid][k] - m); s += e[k]; }
    for (int k = 0; k < S_; ++k) att[tid][k] = e[k] / s;
  }
  __syncthreads();
  if (tid < S_ * HD_) {
    const int qi = tid / HD_, d = tid % HD_;
    float sum = 0.0f;
    for (int k = 0; k < S_; ++k)
      sum += att[qi][k] * base[k * (3 * D_) + 2 * D_ + h * HD_ + d];
    ctx[((size_t)b * S_ + qi) * D_ + h * HD_ + d] = sum;
  }
}

// Stage 1: tPad[0]=LN(x[0][0]+agg_noun, ln3); sPad[j-1]=LN(x[0][j]+agg_verb, ln1)
__global__ void __launch_bounds__(256) ln_stage1_kernel(
    const float* __restrict__ x, const float* __restrict__ aggv,
    const float* __restrict__ aggn,
    const float* __restrict__ l1w, const float* __restrict__ l1b,
    const float* __restrict__ l3w, const float* __restrict__ l3b,
    float* __restrict__ sPad, float* __restrict__ tPad)
{
  const int row = blockIdx.x;                 // 0..6
  const float* xr  = x + row * D_;            // x[0][row]
  const float* add = (row == 0) ? aggn : aggv;
  const float* w   = (row == 0) ? l3w : l1w;
  const float* bb  = (row == 0) ? l3b : l1b;
  float* out = (row == 0) ? tPad : (sPad + (row - 1) * D_);
  __shared__ float red[256];
  const int tid = threadIdx.x;
  float v0 = xr[tid] + add[tid];
  float v1 = xr[tid + 256] + add[tid + 256];
  red[tid] = v0 + v1; __syncthreads();
  for (int s = 128; s > 0; s >>= 1) { if (tid < s) red[tid] += red[tid + s]; __syncthreads(); }
  const float mu = red[0] * (1.0f / D_); __syncthreads();
  const float d0 = v0 - mu, d1 = v1 - mu;
  red[tid] = d0 * d0 + d1 * d1; __syncthreads();
  for (int s = 128; s > 0; s >>= 1) { if (tid < s) red[tid] += red[tid + s]; __syncthreads(); }
  const float rstd = rsqrtf(red[0] * (1.0f / D_) + 1e-5f);
  out[tid]       = d0 * rstd * w[tid]       + bb[tid];
  out[tid + 256] = d1 * rstd * w[tid + 256] + bb[tid + 256];
}

// Stage 2: x[0][0]=LN(tPad[0]+fft, ln4); x[0][j]=LN(sPad[j-1]+ffs[j-1], ln2)
__global__ void __launch_bounds__(256) ln_stage2_kernel(
    const float* __restrict__ sPad, const float* __restrict__ tPad,
    const float* __restrict__ ffs, const float* __restrict__ fft,
    const float* __restrict__ l2w, const float* __restrict__ l2b,
    const float* __restrict__ l4w, const float* __restrict__ l4b,
    float* __restrict__ x)
{
  const int row = blockIdx.x;
  const float* base = (row == 0) ? tPad : (sPad + (row - 1) * D_);
  const float* add  = (row == 0) ? fft  : (ffs  + (row - 1) * D_);
  const float* w    = (row == 0) ? l4w : l2w;
  const float* bb   = (row == 0) ? l4b : l2b;
  __shared__ float red[256];
  const int tid = threadIdx.x;
  float v0 = base[tid] + add[tid];
  float v1 = base[tid + 256] + add[tid + 256];
  red[tid] = v0 + v1; __syncthreads();
  for (int s = 128; s > 0; s >>= 1) { if (tid < s) red[tid] += red[tid + s]; __syncthreads(); }
  const float mu = red[0] * (1.0f / D_); __syncthreads();
  const float d0 = v0 - mu, d1 = v1 - mu;
  red[tid] = d0 * d0 + d1 * d1; __syncthreads();
  for (int s = 128; s > 0; s >>= 1) { if (tid < s) red[tid] += red[tid + s]; __syncthreads(); }
  const float rstd = rsqrtf(red[0] * (1.0f / D_) + 1e-5f);
  float* out = x + row * D_;
  out[tid]       = d0 * rstd * w[tid]       + bb[tid];
  out[tid + 256] = d1 * rstd * w[tid + 256] + bb[tid + 256];
}

__global__ void __launch_bounds__(256) copy_kernel(
    float* __restrict__ dst, const float* __restrict__ src, int n)
{
  const int i = blockIdx.x * blockDim.x + threadIdx.x;
  if (i < n) dst[i] = src[i];
}

// ---------------------------------------------------------------------------
static inline void run_gemm(const float* A, int lda, const float* W, int ldw,
                            const float* bias, float* C, float* part,
                            int MT, int N, int K, int kchunk, int act,
                            float* verbg, hipStream_t stream)
{
  const int ntiles = N / 16;
  const int KS = (K + kchunk - 1) / kchunk;
  dim3 grid(ntiles, KS);
  if (MT == 3)
    wmma_gemm_splitk<3><<<grid, 32, 0, stream>>>(A, lda, W, ldw, part, N, K, kchunk);
  else
    wmma_gemm_splitk<1><<<grid, 32, 0, stream>>>(A, lda, W, ldw, part, N, K, kchunk);
  const int MN = MT * 16 * N;
  reduce_act_kernel<<<(MN + 255) / 256, 256, 0, stream>>>(part, bias, C, MN, N, KS,
                                                          act, verbg);
}

extern "C" void kernel_launch(void* const* d_in, const int* in_sizes, int n_in,
                              void* d_out, int out_size, void* d_ws, size_t ws_size,
                              hipStream_t stream)
{
  const float* features = (const float*)d_in[0];
  const float* role     = (const float*)d_in[1];
  const float* in_w  = (const float*)d_in[2];
  const float* in_b  = (const float*)d_in[3];
  const float* out_w = (const float*)d_in[4];
  const float* out_b = (const float*)d_in[5];
  const float* ln1w  = (const float*)d_in[6];
  const float* ln1b  = (const float*)d_in[7];
  const float* ln2w  = (const float*)d_in[8];
  const float* ln2b  = (const float*)d_in[9];
  const float* ln3w  = (const float*)d_in[10];
  const float* ln3b  = (const float*)d_in[11];
  const float* ln4w  = (const float*)d_in[12];
  const float* ln4b  = (const float*)d_in[13];
  const float* f1w1  = (const float*)d_in[14];
  const float* f1b1  = (const float*)d_in[15];
  const float* f1w2  = (const float*)d_in[16];
  const float* f1b2  = (const float*)d_in[17];
  const float* f2w1  = (const float*)d_in[18];
  const float* f2b1  = (const float*)d_in[19];
  const float* f2w2  = (const float*)d_in[20];
  const float* f2b2  = (const float*)d_in[21];
  const float* a1w   = (const float*)d_in[22];
  const float* a1b   = (const float*)d_in[23];
  const float* a2w   = (const float*)d_in[24];
  const float* a2b   = (const float*)d_in[25];

  // Workspace layout (floats); regions sized so padded-row reads stay inside
  // their own region. Total ~645K floats (~2.6 MB).
  float* ws = (float*)d_ws;
  float* x      = ws;              // 21504   (B,S,D)
  float* src48  = ws + 21504;      // 24576   (48 x 512, rows 42..47 pad)
  float* qkv48  = ws + 46080;      // 73728   (48 x 1536)
  float* ctx48  = ws + 119808;     // 24576   (48 x 512)
  float* msg48  = ws + 144384;     // 50176   (48 x 512 + slack for M=1/K=3072 pad reads)
  float* verb16 = ws + 194560;     // 40960   (16 x 2560, row 0 real)
  float* aggv16 = ws + 235520;     // 8192    (16 x 512, row 0 real)
  float* aggn16 = ws + 243712;     // 8192
  float* sPad   = ws + 251904;     // 8192    (16 x 512, rows 0..5 real)
  float* tPad   = ws + 260096;     // 8192    (16 x 512, row 0 real)
  float* h_s16  = ws + 268288;     // 32768   (16 x 2048)
  float* h_t16  = ws + 301056;     // 32768
  float* ffs16  = ws + 333824;     // 8192
  float* fft16  = ws + 342016;     // 8192
  float* part   = ws + 350208;     // up to 294912 (split-K partials, KS<=12)

  copy_kernel<<<(21504 + 255) / 256, 256, 0, stream>>>(x, features, 21504);

  for (int l = 0; l < L_; ++l) {
    const float* iw  = in_w  + (size_t)l * 3 * D_ * D_;
    const float* ib  = in_b  + (size_t)l * 3 * D_;
    const float* ow  = out_w + (size_t)l * D_ * D_;
    const float* ob  = out_b + (size_t)l * D_;
    const float* w11 = f1w1 + (size_t)l * FF_ * D_;
    const float* b11 = f1b1 + (size_t)l * FF_;
    const float* w12 = f1w2 + (size_t)l * D_ * FF_;
    const float* b12 = f1b2 + (size_t)l * D_;
    const float* w21 = f2w1 + (size_t)l * FF_ * D_;
    const float* b21 = f2b1 + (size_t)l * FF_;
    const float* w22 = f2w2 + (size_t)l * D_ * FF_;
    const float* b22 = f2b2 + (size_t)l * D_;
    const float* aw1 = a1w + (size_t)l * D_ * (5 * D_);
    const float* ab1 = a1b + (size_t)l * D_;
    const float* aw2 = a2w + (size_t)l * D_ * (6 * D_);
    const float* ab2 = a2b + (size_t)l * D_;
    const float* L1w = ln1w + l * D_; const float* L1b = ln1b + l * D_;
    const float* L2w = ln2w + l * D_; const float* L2b = ln2b + l * D_;
    const float* L3w = ln3w + l * D_; const float* L3b = ln3b + l * D_;
    const float* L4w = ln4w + l * D_; const float* L4b = ln4b + l * D_;

    // src = x ; src[:,1:,:] += role_tokens
    add_role_kernel<<<(21504 + 255) / 256, 256, 0, stream>>>(src48, x, role);

    // qkv = src @ in_w^T + in_b   (48pad x 1536, K=512, 384 split-K waves)
    run_gemm(src48, D_, iw, D_, ib, qkv48, part, 3, 3 * D_, D_, 128, 0, nullptr, stream);

    // attention -> ctx (writes rows 0..41)
    attn_kernel<<<B_ * H_, 512, 0, stream>>>(qkv48, ctx48);

    // msg = ctx @ out_w^T + out_b (48pad x 512, K=512); reduce also gathers
    // verb16 row 0 = vec(msg[1:,0,:])
    run_gemm(ctx48, D_, ow, D_, ob, msg48, part, 3, D_, D_, 128, 0, verb16, stream);

    // agg_verb = sigmoid(vec(msg[1:,0,:]) @ a1_w^T + a1_b)   K = 2560
    run_gemm(verb16, 5 * D_, aw1, 5 * D_, ab1, aggv16, part, 1, D_, 5 * D_, 256, 2,
             nullptr, stream);

    // agg_noun = sigmoid(vec(msg[0,1:,:]) @ a2_w^T + a2_b)   K = 3072 (contiguous)
    run_gemm(msg48 + D_, 6 * D_, aw2, 6 * D_, ab2, aggn16, part, 1, D_, 6 * D_, 256, 2,
             nullptr, stream);

    // post-residual LayerNorms -> sPad rows 0..5 (ln1), tPad row 0 (ln3)
    ln_stage1_kernel<<<7, 256, 0, stream>>>(x, aggv16, aggn16,
                                            L1w, L1b, L3w, L3b, sPad, tPad);

    // s-FFN: h = relu(s @ w11^T + b11); ffs = h @ w12^T + b12
    run_gemm(sPad, D_, w11, D_, b11, h_s16, part, 1, FF_, D_, 128, 1, nullptr, stream);
    run_gemm(h_s16, FF_, w12, FF_, b12, ffs16, part, 1, D_, FF_, 256, 0, nullptr, stream);

    // t-FFN
    run_gemm(tPad, D_, w21, D_, b21, h_t16, part, 1, FF_, D_, 128, 1, nullptr, stream);
    run_gemm(h_t16, FF_, w22, FF_, b22, fft16, part, 1, D_, FF_, 256, 0, nullptr, stream);

    // x[0] = LN(residual + ffn, ln2/ln4)
    ln_stage2_kernel<<<7, 256, 0, stream>>>(sPad, tPad, ffs16, fft16,
                                            L2w, L2b, L4w, L4b, x);
  }

  // out = x[0:1]  (1 x S x D)
  copy_kernel<<<(S_ * D_ + 255) / 256, 256, 0, stream>>>((float*)d_out, x, S_ * D_);
}
